// AdaCare_79860621902248
// MI455X (gfx1250) — matrix operand order repaired and checked by reading, not compiled, not run on hardware
//
#include <hip/hip_runtime.h>

// ---------------------------------------------------------------------------
// Types & helpers for bf16 WMMA (gfx1250 / CDNA5, wave32)
// ---------------------------------------------------------------------------
typedef __attribute__((ext_vector_type(16))) __bf16          v16bf;
typedef __attribute__((ext_vector_type(8)))  float           v8f;
typedef __attribute__((ext_vector_type(8)))  unsigned short  u16x8;

union Frag {
    v16bf v;
    u16x8 h[2];
};

__device__ __forceinline__ unsigned short f2bf(float f) {
    unsigned int u = __float_as_uint(f);
    unsigned int r = u + 0x7FFFu + ((u >> 16) & 1u);   // round-to-nearest-even
    return (unsigned short)(r >> 16);
}
__device__ __forceinline__ float bf2f(unsigned short h) {
    return __uint_as_float(((unsigned int)h) << 16);
}
__device__ __forceinline__ float sigmoidf(float x) { return 1.f / (1.f + __expf(-x)); }

// A operand: 16x32 (MxK) bf16.  Base pointer already includes row*lda + hf*8.
// lane(0..15): M=lane, K=[0..7]+[16..23]; lane(16..31): M=lane-16, K=[8..15]+[24..31]
__device__ __forceinline__ Frag load_a_base(const unsigned short* __restrict__ p, int k0) {
    Frag f;
    f.h[0] = *(const u16x8*)(p + k0);
    f.h[1] = *(const u16x8*)(p + k0 + 16);
    return f;
}
// B operand: 32x16 (KxN) bf16 from weight W stored (N rows, K cols) row-major.
// Base pointer already includes nrow*ldw + hf*8.  K=[0..7]+[16..23] / [8..15]+[24..31].
__device__ __forceinline__ Frag load_b_base(const unsigned short* __restrict__ p, int k0) {
    Frag f;
    f.h[0] = *(const u16x8*)(p + k0);
    f.h[1] = *(const u16x8*)(p + k0 + 16);
    return f;
}
__device__ __forceinline__ v8f wmma_bf16(const Frag& a, const Frag& b, v8f c) {
    return __builtin_amdgcn_wmma_f32_16x16x32_bf16(false, a.v, false, b.v,
                                                   (short)0, c, false, false);
}

// ---------------------------------------------------------------------------
// Problem constants
// ---------------------------------------------------------------------------
#define BB   128
#define VV   64
#define NN   32
#define DD   128
#define C_IN 4096
#define KK   64
#define CCC  192
#define HH   256
#define ROWS (BB * VV)          // 8192
#define RNNC (CCC + C_IN)       // 4288

// ---------------------------------------------------------------------------
// Embedding gather -> x bf16 (ROWS x C_IN)
// ---------------------------------------------------------------------------
__global__ void gather_k(const int* __restrict__ ids, const float* __restrict__ emb,
                         unsigned short* __restrict__ xh) {
    int t = blockIdx.x * blockDim.x + threadIdx.x;      // ROWS*NN*32 threads
    int d4 = t & 31; int n = (t >> 5) & 31; int r = t >> 10;
    if (r >= ROWS) return;
    int id = ids[(size_t)r * NN + n];
    const float* e = emb + (size_t)id * DD + d4 * 4;
    unsigned short* o = xh + (size_t)r * C_IN + n * DD + d4 * 4;
    o[0] = f2bf(e[0]); o[1] = f2bf(e[1]); o[2] = f2bf(e[2]); o[3] = f2bf(e[3]);
}

// ---------------------------------------------------------------------------
// Weight conversion / padding / zero fill
// ---------------------------------------------------------------------------
__global__ void conv_w_split_k(const float* __restrict__ w,
                               unsigned short* __restrict__ t0,
                               unsigned short* __restrict__ t1) {
    int i = blockIdx.x * blockDim.x + threadIdx.x;      // KK*C_IN
    if (i >= KK * C_IN) return;
    t0[i] = f2bf(w[2 * i]);
    t1[i] = f2bf(w[2 * i + 1]);
}
__global__ void cvt_pad_k(const float* __restrict__ src, unsigned short* __restrict__ dst,
                          int rows, int cols, int padRows, int padCols) {
    int i = blockIdx.x * blockDim.x + threadIdx.x;
    if (i >= padRows * padCols) return;
    int r = i / padCols, c = i % padCols;
    dst[i] = (r < rows && c < cols) ? f2bf(src[(size_t)r * cols + c]) : (unsigned short)0;
}
__global__ void zero_u16_k(unsigned short* __restrict__ p, int n) {
    int i = blockIdx.x * blockDim.x + threadIdx.x;
    if (i < n) p[i] = 0;
}

// ---------------------------------------------------------------------------
// Cumulative prefix-mean along time (per b, per channel) : bf16 in/out
// ---------------------------------------------------------------------------
__global__ void cummean_k(const unsigned short* __restrict__ src,
                          unsigned short* __restrict__ dst, int C) {
    int idx = blockIdx.x * blockDim.x + threadIdx.x;    // BB*C threads
    int b = idx / C, c = idx % C;
    if (b >= BB) return;
    float s = 0.f;
    size_t base = (size_t)b * VV * C + c;
    for (int v = 0; v < VV; v++) {
        s += bf2f(src[base + (size_t)v * C]);
        dst[base + (size_t)v * C] = f2bf(s / (float)(v + 1));
    }
}

// ---------------------------------------------------------------------------
// Generic bf16 WMMA GEMM:  C(MxN) = A(MxK) * W(NxK)^T  + epilogue
//   EPI 0: out_f32 = acc + bias
//   EPI 1: out_bf16 = relu(acc + bias)
//   EPI 2: out_bf16 = gate_bf16 * sigmoid(acc + bias)
// Block: 256 thr = 8 waves (4 m x 2 n); wave tile 32x64; block tile 128x128.
// K must be a multiple of 64 (software-pipelined ping-pong over 32-wide steps).
// ---------------------------------------------------------------------------
template <int EPI>
__global__ void __launch_bounds__(256)
gemm_bf16_k(const unsigned short* __restrict__ A, int lda,
            const unsigned short* __restrict__ W, int ldw,
            int M, int N, int K,
            const float* __restrict__ bias, int nbias,
            const unsigned short* __restrict__ gate, int ldg, int gcol,
            float* __restrict__ outf, unsigned short* __restrict__ outh,
            int ldo, int ocol) {
    int lane = threadIdx.x & 31;
    int wid  = threadIdx.x >> 5;
    int hf   = lane >> 4, lr = lane & 15;
    int wm = wid & 3, wn = wid >> 2;
    int bm0 = blockIdx.x * 128 + wm * 32;
    int bn0 = blockIdx.y * 128 + wn * 64;
    if (bm0 >= M || bn0 >= N) return;

    const unsigned short* arow0 = A + (size_t)(bm0 + lr) * lda + hf * 8;
    const unsigned short* arow1 = A + (size_t)(bm0 + 16 + lr) * lda + hf * 8;
    const unsigned short* wrow[4];
    for (int j = 0; j < 4; j++)
        wrow[j] = W + (size_t)(bn0 + j * 16 + lr) * ldw + hf * 8;

    v8f acc[2][4];
    v8f zero = {};
    for (int i = 0; i < 2; i++) for (int j = 0; j < 4; j++) acc[i][j] = zero;

    // software-pipelined ping-pong: compute chunk c while loading chunk n
    Frag ca0 = load_a_base(arow0, 0);
    Frag ca1 = load_a_base(arow1, 0);
    Frag cb[4];
    for (int j = 0; j < 4; j++) cb[j] = load_b_base(wrow[j], 0);

    for (int k0 = 0; k0 < K; k0 += 64) {
        Frag na0 = load_a_base(arow0, k0 + 32);
        Frag na1 = load_a_base(arow1, k0 + 32);
        Frag nb[4];
        for (int j = 0; j < 4; j++) nb[j] = load_b_base(wrow[j], k0 + 32);
        for (int j = 0; j < 4; j++) {
            acc[0][j] = wmma_bf16(ca0, cb[j], acc[0][j]);
            acc[1][j] = wmma_bf16(ca1, cb[j], acc[1][j]);
        }
        if (k0 + 64 < K) {
            ca0 = load_a_base(arow0, k0 + 64);
            ca1 = load_a_base(arow1, k0 + 64);
            for (int j = 0; j < 4; j++) cb[j] = load_b_base(wrow[j], k0 + 64);
        }
        for (int j = 0; j < 4; j++) {
            acc[0][j] = wmma_bf16(na0, nb[j], acc[0][j]);
            acc[1][j] = wmma_bf16(na1, nb[j], acc[1][j]);
        }
    }

    for (int i = 0; i < 2; i++) {
        for (int j = 0; j < 4; j++) {
            int col = bn0 + j * 16 + lr;
            float bi = (col < nbias) ? bias[col] : 0.f;
            for (int e = 0; e < 8; e++) {
                int row = bm0 + i * 16 + e + hf * 8;
                float val = acc[i][j][e] + bi;
                if (EPI == 0) {
                    outf[(size_t)row * ldo + ocol + col] = val;
                } else if (EPI == 1) {
                    outh[(size_t)row * ldo + ocol + col] = f2bf(fmaxf(val, 0.f));
                } else {
                    float s = sigmoidf(val);
                    float g = bf2f(gate[(size_t)row * ldg + gcol + col]);
                    outh[(size_t)row * ldo + ocol + col] = f2bf(g * s);
                }
            }
        }
    }
}

// ---------------------------------------------------------------------------
// Causal dilated conv as dual GEMM: Y[r] = X[r]*W1^T + X[r-dil]*W0^T (+b, relu)
// N = 64, block = 8 waves stacked in M (block tile 256x64).
// Lanes whose shifted row falls before the batch start read a zero guard row,
// keeping all loads unconditional (EXEC stays all-ones around the WMMAs).
// ---------------------------------------------------------------------------
__global__ void __launch_bounds__(256)
conv_gemm_k(const unsigned short* __restrict__ X,
            const unsigned short* __restrict__ W0,
            const unsigned short* __restrict__ W1,
            const float* __restrict__ bias,
            const unsigned short* __restrict__ zrow,   // C_IN zeros
            unsigned short* __restrict__ conv,         // ROWS x CCC bf16
            int dil, int ocol) {
    int lane = threadIdx.x & 31, wid = threadIdx.x >> 5;
    int hf = lane >> 4, lr = lane & 15;
    int bm0 = blockIdx.x * 256 + wid * 32;
    int r0 = bm0 + lr, r1 = bm0 + 16 + lr;
    bool s0ok = (r0 & (VV - 1)) >= dil;
    bool s1ok = (r1 & (VV - 1)) >= dil;

    const unsigned short* a0b = X + (size_t)r0 * C_IN + hf * 8;
    const unsigned short* a1b = X + (size_t)r1 * C_IN + hf * 8;
    const unsigned short* s0b = (s0ok ? X + (size_t)(r0 - dil) * C_IN : zrow) + hf * 8;
    const unsigned short* s1b = (s1ok ? X + (size_t)(r1 - dil) * C_IN : zrow) + hf * 8;
    const unsigned short* w0b[4];
    const unsigned short* w1b[4];
    for (int j = 0; j < 4; j++) {
        w0b[j] = W0 + (size_t)(j * 16 + lr) * C_IN + hf * 8;
        w1b[j] = W1 + (size_t)(j * 16 + lr) * C_IN + hf * 8;
    }

    v8f acc[2][4];
    v8f zero = {};
    for (int i = 0; i < 2; i++) for (int j = 0; j < 4; j++) acc[i][j] = zero;

    for (int k0 = 0; k0 < C_IN; k0 += 32) {
        Frag a0 = load_a_base(a0b, k0);
        Frag a1 = load_a_base(a1b, k0);
        Frag s0 = load_a_base(s0b, k0);
        Frag s1 = load_a_base(s1b, k0);
        Frag b0[4], b1[4];
        for (int j = 0; j < 4; j++) {
            b1[j] = load_b_base(w1b[j], k0);
            b0[j] = load_b_base(w0b[j], k0);
        }
        for (int j = 0; j < 4; j++) {
            acc[0][j] = wmma_bf16(a0, b1[j], acc[0][j]);
            acc[0][j] = wmma_bf16(s0, b0[j], acc[0][j]);
            acc[1][j] = wmma_bf16(a1, b1[j], acc[1][j]);
            acc[1][j] = wmma_bf16(s1, b0[j], acc[1][j]);
        }
    }
    for (int i = 0; i < 2; i++) {
        for (int j = 0; j < 4; j++) {
            int col = j * 16 + lr;
            float bi = bias[col];
            for (int e = 0; e < 8; e++) {
                int row = bm0 + i * 16 + e + hf * 8;
                conv[(size_t)row * CCC + ocol + col] = f2bf(fmaxf(acc[i][j][e] + bi, 0.f));
            }
        }
    }
}

// ---------------------------------------------------------------------------
// last valid visit index per batch
// ---------------------------------------------------------------------------
__global__ void last_k(const unsigned char* __restrict__ mask, int* __restrict__ last) {
    int b = blockIdx.x * blockDim.x + threadIdx.x;
    if (b >= BB) return;
    int cnt = 0;
    for (int v = 0; v < VV; v++) {
        bool allm = true;
        for (int n = 0; n < NN; n++) allm = allm && (mask[((size_t)b * VV + v) * NN + n] != 0);
        cnt += allm ? 0 : 1;
    }
    last[b] = cnt - 1;
}

// ---------------------------------------------------------------------------
// Persistent single-workgroup GRU: 64 sequential steps.
//   gh = h(128x256,bf16) @ w_hh(768x256)^T  via WMMA (384 tiles / 32 waves)
//   then fused gate update; h kept in global (L2) as both f32 and bf16-A-layout.
// ---------------------------------------------------------------------------
__global__ void __launch_bounds__(1024)
gru_k(const unsigned short* __restrict__ whh,   // 768x256 bf16
      const float* __restrict__ bhh,            // 768
      const float* __restrict__ gi,             // ROWS x 768 (r = b*64+v), includes b_ih
      const int* __restrict__ last,
      float* __restrict__ h, unsigned short* __restrict__ hb,
      float* __restrict__ gh, float* __restrict__ hsel) {
    int tid = threadIdx.x;
    int lane = tid & 31, wid = tid >> 5;
    int hf = lane >> 4, lr = lane & 15;

    for (int e = tid; e < BB * HH; e += 1024) { h[e] = 0.f; hb[e] = 0; }
    __threadfence();
    __syncthreads();

    for (int v = 0; v < VV; v++) {
        // --- gh = hb @ whh^T ----------------------------------------------
        for (int t = wid * 12; t < wid * 12 + 12; t++) {     // 384 tiles total
            int mt = t & 7, nt = t >> 3;
            int m0 = mt * 16, n0 = nt * 16;
            const unsigned short* ab = hb + (size_t)(m0 + lr) * HH + hf * 8;
            const unsigned short* wb = whh + (size_t)(n0 + lr) * HH + hf * 8;
            v8f acc = {};
            for (int k0 = 0; k0 < HH; k0 += 32) {
                Frag a = load_a_base(ab, k0);
                Frag b = load_b_base(wb, k0);
                acc = wmma_bf16(a, b, acc);
            }
            for (int e = 0; e < 8; e++)
                gh[(size_t)(m0 + e + hf * 8) * 768 + n0 + lr] = acc[e];
        }
        __threadfence();
        __syncthreads();
        // --- fused gate update --------------------------------------------
        for (int e = tid; e < BB * HH; e += 1024) {
            int bb = e >> 8, j = e & 255;
            size_t rg = (size_t)(bb * VV + v) * 768;
            float gir = gi[rg + j];
            float giz = gi[rg + 256 + j];
            float gin = gi[rg + 512 + j];
            float ghr = gh[(size_t)bb * 768 + j]       + bhh[j];
            float ghz = gh[(size_t)bb * 768 + 256 + j] + bhh[256 + j];
            float ghn = gh[(size_t)bb * 768 + 512 + j] + bhh[512 + j];
            float rr = sigmoidf(gir + ghr);
            float zz = sigmoidf(giz + ghz);
            float nn = tanhf(gin + rr * ghn);
            float hn = (1.f - zz) * nn + zz * h[e];
            h[e] = hn;
            hb[e] = f2bf(hn);
            if (last[bb] == v) hsel[e] = hn;
        }
        __threadfence();
        __syncthreads();
    }
}

// ---------------------------------------------------------------------------
// Output head: out[b,o] = out_w[o,:] . hsel[b,:] + out_b[o]
// ---------------------------------------------------------------------------
__global__ void out_k(const float* __restrict__ hsel, const float* __restrict__ ow,
                      const float* __restrict__ ob, float* __restrict__ out) {
    int t = blockIdx.x * blockDim.x + threadIdx.x;
    if (t >= BB * 2) return;
    int b = t >> 1, o = t & 1;
    float s = ob[o];
    for (int j = 0; j < HH; j++) s += ow[o * HH + j] * hsel[(size_t)b * HH + j];
    out[b * 2 + o] = s;
}

// ---------------------------------------------------------------------------
// Host orchestration
// ---------------------------------------------------------------------------
extern "C" void kernel_launch(void* const* d_in, const int* in_sizes, int n_in,
                              void* d_out, int out_size, void* d_ws, size_t ws_size,
                              hipStream_t stream) {
    (void)in_sizes; (void)n_in; (void)out_size;
    const int*           node_ids  = (const int*)d_in[0];
    const unsigned char* attn_mask = (const unsigned char*)d_in[1];
    const float* node_embed = (const float*)d_in[6];
    const float* conv1_w = (const float*)d_in[7];  const float* conv1_b = (const float*)d_in[8];
    const float* conv3_w = (const float*)d_in[9];  const float* conv3_b = (const float*)d_in[10];
    const float* conv5_w = (const float*)d_in[11]; const float* conv5_b = (const float*)d_in[12];
    const float* cse_c_w = (const float*)d_in[13]; const float* cse_c_b = (const float*)d_in[14];
    const float* cse_r_w = (const float*)d_in[15]; const float* cse_r_b = (const float*)d_in[16];
    const float* ise_c_w = (const float*)d_in[17]; const float* ise_c_b = (const float*)d_in[18];
    const float* ise_r_w = (const float*)d_in[19]; const float* ise_r_b = (const float*)d_in[20];
    const float* w_ih = (const float*)d_in[21];    const float* b_ih = (const float*)d_in[22];
    const float* w_hh = (const float*)d_in[23];    const float* b_hh = (const float*)d_in[24];
    const float* out_w = (const float*)d_in[25];   const float* out_b = (const float*)d_in[26];

    char* ws = (char*)d_ws;
    size_t off = 0;
    auto take = [&](size_t bytes) -> void* {
        void* p = ws + off;
        off = (off + bytes + 255) & ~(size_t)255;
        return p;
    };
    unsigned short* xh    = (unsigned short*)take((size_t)ROWS * C_IN * 2);
    unsigned short* cmx   = (unsigned short*)take((size_t)ROWS * C_IN * 2);
    unsigned short* convb = (unsigned short*)take((size_t)ROWS * CCC * 2);
    unsigned short* cmc   = (unsigned short*)take((size_t)ROWS * CCC * 2);
    unsigned short* seC   = (unsigned short*)take((size_t)ROWS * 64 * 2);
    unsigned short* seX   = (unsigned short*)take((size_t)ROWS * 1024 * 2);
    unsigned short* rnn   = (unsigned short*)take((size_t)ROWS * RNNC * 2);
    float*          gi    = (float*)take((size_t)ROWS * 768 * 4);
    unsigned short* wc1t0 = (unsigned short*)take((size_t)KK * C_IN * 2);
    unsigned short* wc1t1 = (unsigned short*)take((size_t)KK * C_IN * 2);
    unsigned short* wc3t0 = (unsigned short*)take((size_t)KK * C_IN * 2);
    unsigned short* wc3t1 = (unsigned short*)take((size_t)KK * C_IN * 2);
    unsigned short* wc5t0 = (unsigned short*)take((size_t)KK * C_IN * 2);
    unsigned short* wc5t1 = (unsigned short*)take((size_t)KK * C_IN * 2);
    unsigned short* wCseC = (unsigned short*)take((size_t)64 * CCC * 2);    // padded 48->64 rows
    unsigned short* wCseR = (unsigned short*)take((size_t)CCC * 64 * 2);    // padded 48->64 cols
    unsigned short* wIseC = (unsigned short*)take((size_t)1024 * C_IN * 2);
    unsigned short* wIseR = (unsigned short*)take((size_t)C_IN * 1024 * 2);
    unsigned short* wIH   = (unsigned short*)take((size_t)768 * RNNC * 2);
    unsigned short* wHH   = (unsigned short*)take((size_t)768 * HH * 2);
    unsigned short* zrow  = (unsigned short*)take((size_t)C_IN * 2);
    float*          ghb   = (float*)take((size_t)BB * 768 * 4);
    float*          hbuf  = (float*)take((size_t)BB * HH * 4);
    unsigned short* hbf   = (unsigned short*)take((size_t)BB * HH * 2);
    float*          hsel  = (float*)take((size_t)BB * HH * 4);
    int*            lastb = (int*)take((size_t)BB * 4);
    if (off > ws_size) return;   // workspace too small: refuse (deterministic)

    dim3 thr(256);
    // ---- weight conversion -------------------------------------------------
    conv_w_split_k<<<(KK * C_IN + 255) / 256, thr, 0, stream>>>(conv1_w, wc1t0, wc1t1);
    conv_w_split_k<<<(KK * C_IN + 255) / 256, thr, 0, stream>>>(conv3_w, wc3t0, wc3t1);
    conv_w_split_k<<<(KK * C_IN + 255) / 256, thr, 0, stream>>>(conv5_w, wc5t0, wc5t1);
    cvt_pad_k<<<(64 * CCC + 255) / 256, thr, 0, stream>>>(cse_c_w, wCseC, 48, CCC, 64, CCC);
    cvt_pad_k<<<(CCC * 64 + 255) / 256, thr, 0, stream>>>(cse_r_w, wCseR, CCC, 48, CCC, 64);
    cvt_pad_k<<<(1024 * C_IN + 255) / 256, thr, 0, stream>>>(ise_c_w, wIseC, 1024, C_IN, 1024, C_IN);
    cvt_pad_k<<<(C_IN * 1024 + 255) / 256, thr, 0, stream>>>(ise_r_w, wIseR, C_IN, 1024, C_IN, 1024);
    cvt_pad_k<<<(768 * RNNC + 255) / 256, thr, 0, stream>>>(w_ih, wIH, 768, RNNC, 768, RNNC);
    cvt_pad_k<<<(768 * HH + 255) / 256, thr, 0, stream>>>(w_hh, wHH, 768, HH, 768, HH);
    zero_u16_k<<<(C_IN + 255) / 256, thr, 0, stream>>>(zrow, C_IN);

    // ---- gather + last index ----------------------------------------------
    gather_k<<<(ROWS * 1024 + 255) / 256, thr, 0, stream>>>(node_ids, node_embed, xh);
    last_k<<<1, 128, 0, stream>>>(attn_mask, lastb);

    // ---- causal convs (relu fused) -----------------------------------------
    conv_gemm_k<<<ROWS / 256, thr, 0, stream>>>(xh, wc1t0, wc1t1, conv1_b, zrow, convb, 1, 0);
    conv_gemm_k<<<ROWS / 256, thr, 0, stream>>>(xh, wc3t0, wc3t1, conv3_b, zrow, convb, 3, 64);
    conv_gemm_k<<<ROWS / 256, thr, 0, stream>>>(xh, wc5t0, wc5t1, conv5_b, zrow, convb, 5, 128);

    // ---- prefix means -------------------------------------------------------
    cummean_k<<<(BB * C_IN + 255) / 256, thr, 0, stream>>>(xh, cmx, C_IN);
    cummean_k<<<(BB * CCC + 255) / 256, thr, 0, stream>>>(convb, cmc, CCC);

    // ---- conv squeeze-excite -----------------------------------------------
    gemm_bf16_k<1><<<dim3(ROWS / 128, 1), thr, 0, stream>>>(
        cmc, CCC, wCseC, CCC, ROWS, 64, CCC, cse_c_b, 48,
        nullptr, 0, 0, nullptr, seC, 64, 0);
    gemm_bf16_k<2><<<dim3(ROWS / 128, 2), thr, 0, stream>>>(
        seC, 64, wCseR, 64, ROWS, CCC, 64, cse_r_b, CCC,
        convb, CCC, 0, nullptr, rnn, RNNC, 0);

    // ---- input squeeze-excite ----------------------------------------------
    gemm_bf16_k<1><<<dim3(ROWS / 128, 8), thr, 0, stream>>>(
        cmx, C_IN, wIseC, C_IN, ROWS, 1024, C_IN, ise_c_b, 1024,
        nullptr, 0, 0, nullptr, seX, 1024, 0);
    gemm_bf16_k<2><<<dim3(ROWS / 128, 32), thr, 0, stream>>>(
        seX, 1024, wIseR, 1024, ROWS, C_IN, 1024, ise_r_b, C_IN,
        xh, C_IN, 0, nullptr, rnn, RNNC, CCC);

    // ---- GRU input projection (all timesteps at once) ----------------------
    gemm_bf16_k<0><<<dim3(ROWS / 128, 6), thr, 0, stream>>>(
        rnn, RNNC, wIH, RNNC, ROWS, 768, RNNC, b_ih, 768,
        nullptr, 0, 0, gi, nullptr, 768, 0);

    // ---- sequential GRU (single persistent workgroup) ----------------------
    gru_k<<<1, 1024, 0, stream>>>(wHH, b_hh, gi, lastb, hbuf, hbf, ghb, hsel);

    // ---- output head --------------------------------------------------------
    out_k<<<1, 256, 0, stream>>>(hsel, out_w, out_b, (float*)d_out);
}
